// GLBert4Rec_21509196218375
// MI455X (gfx1250) — matrix operand order repaired and checked by reading, not compiled, not run on hardware
//
#include <hip/hip_runtime.h>

#define SEQ    50
#define LP     64     // padded sequence length (4 x 16 tiles)
#define HID    128
#define NHEAD  4
#define HDIM   32
#define INNER  512
#define NLAYER 2
#define NITEMS 50000
#define BSZ    2048

typedef __bf16 bf16;
typedef __attribute__((ext_vector_type(16))) __bf16 v16bf;
typedef __attribute__((ext_vector_type(8)))  float  v8f;

// ---------------------------------------------------------------------------
// WMMA helpers (CDNA5 wave32, D = A(16x32 bf16) * B(32x16 bf16) + C(16x16 f32))
// ---------------------------------------------------------------------------
__device__ __forceinline__ v8f wmma_bf16(v16bf a, v16bf b, v8f c) {
  return __builtin_amdgcn_wmma_f32_16x16x32_bf16(false, a, false, b, (short)0, c,
                                                 false, false);
}

// A tile: 16x32 bf16; lanes 0-15 rows M=0..15 (K 0..7 / 16..23),
// lanes 16-31 same rows (K 8..15 / 24..31). p points at tile origin, ld in elems.
__device__ __forceinline__ v16bf load_a16(const bf16* p, int ld) {
  const int lane = threadIdx.x & 31;
  const bf16* r = p + (lane & 15) * ld + ((lane >> 4) << 3);
  v16bf a;
#pragma unroll
  for (int i = 0; i < 8; ++i) { a[i] = r[i]; a[i + 8] = r[16 + i]; }
  return a;
}

// B tile: 32x16 bf16, row-major source B[k*ld + n]; lane%16 = n,
// lanes 0-15 K 0..7 / 16..23, lanes 16-31 K 8..15 / 24..31.
__device__ __forceinline__ v16bf load_b16(const bf16* p, int ld) {
  const int lane = threadIdx.x & 31;
  const bf16* r = p + ((lane >> 4) << 3) * ld + (lane & 15);
  v16bf b;
#pragma unroll
  for (int i = 0; i < 8; ++i) { b[i] = r[i * ld]; b[i + 8] = r[(16 + i) * ld]; }
  return b;
}

// Tiled GEMM over a 256-thread block (8 waves). A: (Mt*16 x K) bf16 row-major,
// B: (K x Nt*16) bf16 row-major. Each wave owns pairs of ADJACENT column tiles:
// one shared A-tile load feeds two independent WMMA chains (hides the 4-coexec
// accumulator RAW hazard) and the two stores hit the same 128B cacheline.
// bias indexed by output column (may be null). store(mt, nt, acc) commits one
// 16x16 f32 tile (C/D layout: lane=n, vgpr rows m / m+8).
template <int Mt, int Nt, int K, typename StoreF>
__device__ __forceinline__ void gemm_tiles(const bf16* A, int lda, const bf16* B,
                                           int ldb, const float* bias,
                                           StoreF&& store) {
  static_assert(Nt % 2 == 0, "Nt must be even");
  const int wave = __builtin_amdgcn_readfirstlane(threadIdx.x >> 5);  // scalar
  const int lane = threadIdx.x & 31;
  constexpr int NP = Mt * (Nt / 2);
  for (int pp = wave; pp < NP; pp += 8) {
    const int mt = pp / (Nt / 2);
    const int nt = (pp - mt * (Nt / 2)) * 2;
    v8f acc0, acc1;
    const float b0 = bias ? bias[nt * 16 + (lane & 15)] : 0.0f;
    const float b1 = bias ? bias[nt * 16 + 16 + (lane & 15)] : 0.0f;
#pragma unroll
    for (int r = 0; r < 8; ++r) { acc0[r] = b0; acc1[r] = b1; }
#pragma unroll
    for (int k0 = 0; k0 < K; k0 += 32) {
      v16bf a   = load_a16(A + mt * 16 * lda + k0, lda);
      v16bf bb0 = load_b16(B + k0 * ldb + nt * 16, ldb);
      v16bf bb1 = load_b16(B + k0 * ldb + nt * 16 + 16, ldb);
      acc0 = wmma_bf16(a, bb0, acc0);
      acc1 = wmma_bf16(a, bb1, acc1);
    }
    store(mt, nt, acc0);
    store(mt, nt + 1, acc1);
  }
}

// ---------------------------------------------------------------------------
// Weight prep: f32 (R x C) -> bf16 transposed (C x R); B[k][n] = W[n][k]
// ---------------------------------------------------------------------------
__global__ void transpose_bf16(const float* __restrict__ src, bf16* __restrict__ dst,
                               int R, int C) {
  int i = blockIdx.x * blockDim.x + threadIdx.x;
  if (i >= R * C) return;
  int r = i / C, c = i - r * C;
  dst[c * R + r] = (bf16)src[i];
}

// ---------------------------------------------------------------------------
// Fused per-sequence transformer forward (one block per batch element)
// ---------------------------------------------------------------------------
struct FwdParams {
  const int* ids;
  const float* A;
  const float* emb;
  const float* pos;
  const float* mixb;
  const float* bq; const float* bk; const float* bv; const float* bo;
  const float* ln1g; const float* ln1b;
  const float* b1;  const float* b2;
  const float* ln2g; const float* ln2b;
  const float* attW; const float* attb;
  const float* sessW; const float* sessb;
  const bf16* mixWT; const bf16* WqT; const bf16* WkT; const bf16* WvT;
  const bf16* WoT;  const bf16* W1T; const bf16* W2T;
  bf16* sessOut;
};

// Parallel residual + LayerNorm: 4 threads per row, partial sums via red[512].
__device__ __forceinline__ void ln_rows(float* xf, bf16* xb, const float* add,
                                        const float* g, const float* be,
                                        float* red) {
  const int tid = threadIdx.x;
  const int row = tid >> 2, q = tid & 3;
  float* xr = xf + row * HID;
  const float* ar = add + row * HID;
  const int c0 = q * 32;
  float s = 0.f, ss = 0.f;
  for (int d = c0; d < c0 + 32; ++d) {
    float v = xr[d] + ar[d];
    s += v; ss += v * v;
  }
  red[tid] = s;
  red[256 + tid] = ss;
  __syncthreads();
  const int rb = row << 2;
  const float sum = red[rb] + red[rb + 1] + red[rb + 2] + red[rb + 3];
  const float sq  = red[256 + rb] + red[256 + rb + 1] + red[256 + rb + 2] +
                    red[256 + rb + 3];
  const float mean = sum * (1.0f / HID);
  const float var = sq * (1.0f / HID) - mean * mean;
  const float rstd = rsqrtf(var + 1e-5f);
  const bool live = row < SEQ;  // keep padded rows exactly zero
  for (int d = c0; d < c0 + 32; ++d) {
    float y = live ? ((xr[d] + ar[d] - mean) * rstd * g[d] + be[d]) : 0.f;
    xr[d] = y;
    xb[row * HID + d] = (bf16)y;
  }
}

// Parallel masked row softmax over tf (LP x LP) -> pb bf16; 4 threads per row.
__device__ __forceinline__ void softmax_rows(float* tf, bf16* pb, float* red) {
  const int tid = threadIdx.x;
  const int row = tid >> 2, q = tid & 3;
  float* srow = tf + row * LP;
  const int c0 = q * 16;
  float mx = -1e30f;
  for (int c = c0; c < c0 + 16; ++c)
    if (c < SEQ) mx = fmaxf(mx, srow[c]);
  red[tid] = mx;
  __syncthreads();
  const int rb = row << 2;
  mx = fmaxf(fmaxf(red[rb], red[rb + 1]), fmaxf(red[rb + 2], red[rb + 3]));
  float s = 0.f;
  for (int c = c0; c < c0 + 16; ++c) {
    float e = (c < SEQ) ? __expf(srow[c] - mx) : 0.f;
    srow[c] = e;
    s += e;
  }
  red[256 + tid] = s;
  __syncthreads();
  const float tot = red[256 + rb] + red[256 + rb + 1] + red[256 + rb + 2] +
                    red[256 + rb + 3];
  const float inv = 1.f / tot;
  for (int c = c0; c < c0 + 16; ++c)
    pb[row * LP + c] = (bf16)(srow[c] * inv);
}

__global__ __launch_bounds__(256) void fwd_kernel(FwdParams p) {
  extern __shared__ char smem[];
  bf16* xb  = (bf16*)smem;             // LP*HID   current x (bf16)
  bf16* tb  = xb + LP * HID;           // LP*INNER general bf16 temp
  bf16* pb  = tb + LP * INNER;         // LP*LP    softmax probs
  float* xf = (float*)(pb + LP * LP);  // LP*HID   current x (f32)
  float* tf = xf + LP * HID;           // LP*HID   f32 temp (scores / pre-LN)
  float* red = tf + LP * HID;          // 512 floats reduction scratch

  const int b = blockIdx.x;
  const int tid = threadIdx.x;
  const int lane = tid & 31;

  if (tid == 0) {  // warm L2/WGP$ for the weight stream (global_prefetch_b8)
    __builtin_prefetch(p.mixWT, 0, 0);
    __builtin_prefetch(p.W1T, 0, 0);
    __builtin_prefetch(p.W2T, 0, 0);
  }

  // x = emb[ids] + pos (rows >= SEQ zeroed)
  const int* ids = p.ids + b * SEQ;
  for (int idx = tid; idx < LP * HID; idx += 256) {
    int m = idx >> 7, d = idx & (HID - 1);
    float v = 0.f;
    if (m < SEQ) v = p.emb[ids[m] * HID + d] + p.pos[m * HID + d];
    xf[idx] = v;
    xb[idx] = (bf16)v;
  }
  __syncthreads();

  for (int li = 0; li < NLAYER; ++li) {
    // --- load session-graph A as bf16 (cols 0..63: A_in, 64..127: A_out) ---
    for (int idx = tid; idx < LP * 128; idx += 256) {
      int l = idx >> 7, c = idx & 127, m = c & 63;
      float v = 0.f;
      if (l < SEQ && m < SEQ)
        v = p.A[(b * SEQ + l) * (2 * SEQ) + (c >> 6) * SEQ + m];
      tb[idx] = (bf16)v;
    }
    __syncthreads();

    bf16* Ab = tb;
    bf16* cc = tb + LP * HID;  // LP x 256 concat buffer
    // x_in = A_in @ x ; x_out = A_out @ x  -> cc
    gemm_tiles<4, 8, LP>(Ab, 128, xb, HID, nullptr, [&](int mt, int nt, v8f acc) {
      int n = lane & 15, mb = (lane >> 4) << 3;
#pragma unroll
      for (int r = 0; r < 8; ++r)
        cc[(mt * 16 + mb + r) * 256 + nt * 16 + n] = (bf16)acc[r];
    });
    gemm_tiles<4, 8, LP>(Ab + LP, 128, xb, HID, nullptr,
                         [&](int mt, int nt, v8f acc) {
      int n = lane & 15, mb = (lane >> 4) << 3;
#pragma unroll
      for (int r = 0; r < 8; ++r)
        cc[(mt * 16 + mb + r) * 256 + 128 + nt * 16 + n] = (bf16)acc[r];
    });
    __syncthreads();

    // --- mix: x = cc @ mixW^T + mixb ---
    gemm_tiles<4, 8, 256>(cc, 256, p.mixWT + li * 256 * HID, HID,
                          p.mixb + li * HID, [&](int mt, int nt, v8f acc) {
      int n = lane & 15, mb = (lane >> 4) << 3;
#pragma unroll
      for (int r = 0; r < 8; ++r) {
        int o = (mt * 16 + mb + r) * HID + nt * 16 + n;
        xf[o] = acc[r];
        xb[o] = (bf16)acc[r];
      }
    });
    __syncthreads();

    // --- q, k(transposed), v ---
    bf16* qb = tb;
    bf16* kT = tb + LP * HID;      // HID x LP
    bf16* vb = tb + 2 * LP * HID;  // LP x HID
    bf16* ob = tb + 3 * LP * HID;  // LP x HID attention output
    gemm_tiles<4, 8, HID>(xb, HID, p.WqT + li * HID * HID, HID, p.bq + li * HID,
                          [&](int mt, int nt, v8f acc) {
      int n = lane & 15, mb = (lane >> 4) << 3;
#pragma unroll
      for (int r = 0; r < 8; ++r)
        qb[(mt * 16 + mb + r) * HID + nt * 16 + n] = (bf16)acc[r];
    });
    gemm_tiles<4, 8, HID>(xb, HID, p.WkT + li * HID * HID, HID, p.bk + li * HID,
                          [&](int mt, int nt, v8f acc) {
      int n = lane & 15, mb = (lane >> 4) << 3;
#pragma unroll
      for (int r = 0; r < 8; ++r)
        kT[(nt * 16 + n) * LP + mt * 16 + mb + r] = (bf16)acc[r];
    });
    gemm_tiles<4, 8, HID>(xb, HID, p.WvT + li * HID * HID, HID, p.bv + li * HID,
                          [&](int mt, int nt, v8f acc) {
      int n = lane & 15, mb = (lane >> 4) << 3;
#pragma unroll
      for (int r = 0; r < 8; ++r)
        vb[(mt * 16 + mb + r) * HID + nt * 16 + n] = (bf16)acc[r];
    });
    __syncthreads();

    // --- attention, one head at a time ---
    const float scale = 0.17677669529663687f;  // 1/sqrt(32)
    for (int h = 0; h < NHEAD; ++h) {
      gemm_tiles<4, 4, HDIM>(qb + h * HDIM, HID, kT + h * HDIM * LP, LP, nullptr,
                             [&](int mt, int nt, v8f acc) {
        int n = lane & 15, mb = (lane >> 4) << 3;
#pragma unroll
        for (int r = 0; r < 8; ++r)
          tf[(mt * 16 + mb + r) * LP + nt * 16 + n] = acc[r] * scale;
      });
      __syncthreads();
      softmax_rows(tf, pb, red);
      __syncthreads();
      gemm_tiles<4, 2, LP>(pb, LP, vb + h * HDIM, HID, nullptr,
                           [&](int mt, int nt, v8f acc) {
        int n = lane & 15, mb = (lane >> 4) << 3;
#pragma unroll
        for (int r = 0; r < 8; ++r)
          ob[(mt * 16 + mb + r) * HID + h * HDIM + nt * 16 + n] = (bf16)acc[r];
      });
      __syncthreads();
    }

    // --- o @ Wo^T + bo -> tf ; x = LN1(x + o) ---
    gemm_tiles<4, 8, HID>(ob, HID, p.WoT + li * HID * HID, HID, p.bo + li * HID,
                          [&](int mt, int nt, v8f acc) {
      int n = lane & 15, mb = (lane >> 4) << 3;
#pragma unroll
      for (int r = 0; r < 8; ++r)
        tf[(mt * 16 + mb + r) * HID + nt * 16 + n] = acc[r];
    });
    __syncthreads();
    ln_rows(xf, xb, tf, p.ln1g + li * HID, p.ln1b + li * HID, red);
    __syncthreads();

    // --- FFN: relu(x @ W1^T + b1) @ W2^T + b2 ; x = LN2(x + ff) ---
    gemm_tiles<4, 32, HID>(xb, HID, p.W1T + li * HID * INNER, INNER,
                           p.b1 + li * INNER, [&](int mt, int nt, v8f acc) {
      int n = lane & 15, mb = (lane >> 4) << 3;
#pragma unroll
      for (int r = 0; r < 8; ++r) {
        float v = acc[r]; v = v > 0.f ? v : 0.f;
        tb[(mt * 16 + mb + r) * INNER + nt * 16 + n] = (bf16)v;
      }
    });
    __syncthreads();
    gemm_tiles<4, 8, INNER>(tb, INNER, p.W2T + li * INNER * HID, HID,
                            p.b2 + li * HID, [&](int mt, int nt, v8f acc) {
      int n = lane & 15, mb = (lane >> 4) << 3;
#pragma unroll
      for (int r = 0; r < 8; ++r)
        tf[(mt * 16 + mb + r) * HID + nt * 16 + n] = acc[r];
    });
    __syncthreads();
    ln_rows(xf, xb, tf, p.ln2g + li * HID, p.ln2b + li * HID, red);
    __syncthreads();
  }

  // --- session head: attention pooling + concat + sessW ---
  if (tid < LP) {
    float a = 0.f;
    if (tid < SEQ) {
      const float* xr = xf + tid * HID;
      float s = p.attb[0];
      for (int d = 0; d < HID; ++d) s += xr[d] * p.attW[d];
      a = s;
    }
    red[tid] = a;
  }
  __syncthreads();
  if (tid == 0) {
    float s = 0.f;
    for (int l = 0; l < SEQ; ++l) s += red[l];
    red[LP] = s;
  }
  __syncthreads();
  if (tid < HID) {
    float inv = 1.f / red[LP];
    float g = 0.f;
    for (int l = 0; l < SEQ; ++l) g += xf[l * HID + tid] * red[l];
    red[128 + tid] = xf[(SEQ - 1) * HID + tid];  // out_local
    red[256 + tid] = g * inv;                    // out_glob
  }
  __syncthreads();
  if (tid < HID) {
    float s = p.sessb[tid];
    const float* wrow = p.sessW + tid * 2 * HID;
    for (int e = 0; e < 2 * HID; ++e) s += red[128 + e] * wrow[e];
    p.sessOut[b * HID + tid] = (bf16)s;
  }
}

// ---------------------------------------------------------------------------
// logits = sess @ emb[1:]^T  (2048 x 49999, K=128); B block staged in LDS
// ---------------------------------------------------------------------------
__global__ __launch_bounds__(256) void logits_kernel(const bf16* __restrict__ sess,
                                                     const bf16* __restrict__ embT,
                                                     float* __restrict__ out) {
  __shared__ bf16 As[64 * HID];    // 16 KB
  __shared__ bf16 Bs[HID * 128];   // 32 KB
  const int r0 = blockIdx.y * 64;
  const int c0 = blockIdx.x * 128;
  const int tid = threadIdx.x;
  const int lane = tid & 31;
  for (int idx = tid; idx < 64 * HID; idx += 256) As[idx] = sess[r0 * HID + idx];
  for (int idx = tid; idx < HID * 128; idx += 256) {
    int k = idx >> 7, j = idx & 127;
    Bs[idx] = embT[(size_t)k * NITEMS + 1 + c0 + j];
  }
  __syncthreads();
  gemm_tiles<4, 8, HID>(As, HID, Bs, 128, nullptr, [&](int mt, int nt, v8f acc) {
    int n = lane & 15, mb = (lane >> 4) << 3;
#pragma unroll
    for (int r = 0; r < 8; ++r) {
      int col = c0 + nt * 16 + n;
      if (col < NITEMS - 1)
        out[(long long)(r0 + mt * 16 + mb + r) * (NITEMS - 1) + col] = acc[r];
    }
  });
}

// ---------------------------------------------------------------------------
extern "C" void kernel_launch(void* const* d_in, const int* in_sizes, int n_in,
                              void* d_out, int out_size, void* d_ws, size_t ws_size,
                              hipStream_t stream) {
  (void)in_sizes; (void)n_in; (void)out_size; (void)ws_size;
  const int*   ids  = (const int*)d_in[0];
  const float* A    = (const float*)d_in[1];
  const float* emb  = (const float*)d_in[2];
  const float* pos  = (const float*)d_in[3];
  const float* mixW = (const float*)d_in[4];
  const float* mixb = (const float*)d_in[5];
  const float* Wq   = (const float*)d_in[6];
  const float* bq   = (const float*)d_in[7];
  const float* Wk   = (const float*)d_in[8];
  const float* bk   = (const float*)d_in[9];
  const float* Wv   = (const float*)d_in[10];
  const float* bv   = (const float*)d_in[11];
  const float* Wo   = (const float*)d_in[12];
  const float* bo   = (const float*)d_in[13];
  const float* ln1g = (const float*)d_in[14];
  const float* ln1b = (const float*)d_in[15];
  const float* W1   = (const float*)d_in[16];
  const float* b1   = (const float*)d_in[17];
  const float* W2   = (const float*)d_in[18];
  const float* b2   = (const float*)d_in[19];
  const float* ln2g = (const float*)d_in[20];
  const float* ln2b = (const float*)d_in[21];
  const float* attW = (const float*)d_in[22];
  const float* attb = (const float*)d_in[23];
  const float* sessW = (const float*)d_in[24];
  const float* sessb = (const float*)d_in[25];

  // workspace carve-up (bf16 elements)
  bf16* ws = (bf16*)d_ws;
  size_t off = 0;
  bf16* embT  = ws + off; off += (size_t)HID * NITEMS;  // emb^T (128 x 50000)
  bf16* mixWT = ws + off; off += (size_t)NLAYER * 256 * HID;
  bf16* WqT   = ws + off; off += (size_t)NLAYER * HID * HID;
  bf16* WkT   = ws + off; off += (size_t)NLAYER * HID * HID;
  bf16* WvT   = ws + off; off += (size_t)NLAYER * HID * HID;
  bf16* WoT   = ws + off; off += (size_t)NLAYER * HID * HID;
  bf16* W1T   = ws + off; off += (size_t)NLAYER * HID * INNER;
  bf16* W2T   = ws + off; off += (size_t)NLAYER * INNER * HID;
  bf16* sessOut = ws + off; off += (size_t)BSZ * HID;

  auto tg = [](int n) { return (n + 255) / 256; };
  transpose_bf16<<<tg(NITEMS * HID), 256, 0, stream>>>(emb, embT, NITEMS, HID);
  for (int i = 0; i < NLAYER; ++i) {
    transpose_bf16<<<tg(HID * 256), 256, 0, stream>>>(mixW + i * HID * 256,
                                                      mixWT + i * 256 * HID, HID, 256);
    transpose_bf16<<<tg(HID * HID), 256, 0, stream>>>(Wq + i * HID * HID,
                                                      WqT + i * HID * HID, HID, HID);
    transpose_bf16<<<tg(HID * HID), 256, 0, stream>>>(Wk + i * HID * HID,
                                                      WkT + i * HID * HID, HID, HID);
    transpose_bf16<<<tg(HID * HID), 256, 0, stream>>>(Wv + i * HID * HID,
                                                      WvT + i * HID * HID, HID, HID);
    transpose_bf16<<<tg(HID * HID), 256, 0, stream>>>(Wo + i * HID * HID,
                                                      WoT + i * HID * HID, HID, HID);
    transpose_bf16<<<tg(INNER * HID), 256, 0, stream>>>(W1 + i * INNER * HID,
                                                        W1T + i * HID * INNER, INNER, HID);
    transpose_bf16<<<tg(HID * INNER), 256, 0, stream>>>(W2 + i * HID * INNER,
                                                        W2T + i * INNER * HID, HID, INNER);
  }

  FwdParams fp;
  fp.ids = ids; fp.A = A; fp.emb = emb; fp.pos = pos; fp.mixb = mixb;
  fp.bq = bq; fp.bk = bk; fp.bv = bv; fp.bo = bo;
  fp.ln1g = ln1g; fp.ln1b = ln1b; fp.b1 = b1; fp.b2 = b2;
  fp.ln2g = ln2g; fp.ln2b = ln2b; fp.attW = attW; fp.attb = attb;
  fp.sessW = sessW; fp.sessb = sessb;
  fp.mixWT = mixWT; fp.WqT = WqT; fp.WkT = WkT; fp.WvT = WvT; fp.WoT = WoT;
  fp.W1T = W1T; fp.W2T = W2T; fp.sessOut = sessOut;

  const size_t smem = (size_t)(LP * HID + LP * INNER + LP * LP) * sizeof(bf16) +
                      (size_t)(LP * HID * 2 + 512) * sizeof(float);  // ~154 KB
  fwd_kernel<<<BSZ, 256, smem, stream>>>(fp);

  dim3 lg((NITEMS - 1 + 127) / 128, BSZ / 64);
  logits_kernel<<<lg, 256, 0, stream>>>(sessOut, embT, (float*)d_out);
}
